// Model_84146999263727
// MI455X (gfx1250) — compile-verified, matrix operand order checked by reading, not run on hardware
//
#include <hip/hip_runtime.h>
#include <hip/hip_bf16.h>

typedef __attribute__((ext_vector_type(16))) _Float16 v16h;
typedef __attribute__((ext_vector_type(8)))  float    v8f;

constexpr int BATCH = 4;
constexpr int N0    = 4096;
constexpr int NEWP  = 1152;
constexpr int STEPS = 10;
constexpr int CN    = N0 + NEWP * STEPS;   // 15616 final canvas rows
constexpr int GD    = 256;                 // GDIM
constexpr int HD    = 128;                 // HDIM
constexpr int KSEL  = 64;                  // top-K
constexpr int CTXK  = 387;                 // 3 + GDIM + HDIM
constexpr int CTX_KT = 13;                 // ceil(387/32)

// ---------------- fragment layout helpers (CDNA5 16-bit A/B 16x32) --------
// A 16x32 (M x K): lane = grp*16 + m ; VGPR j holds K pair:
//   grp0: j0..3 -> K 0..7, j4..7 -> K 16..23 ; grp1: +8
__device__ __forceinline__ int frag_kk(int lane, int idx) {
  int j   = idx >> 1, bit = idx & 1;
  int jj  = j & 3;
  int hi  = (j >> 2) << 4;            // 0 or 16
  int grp = ((lane >> 4) & 1) << 3;   // 0 or 8
  return hi + grp + jj * 2 + bit;
}

union AFrag { v16h v; _Float16 e[16]; };
union CFrag { v8f  v; float    e[8];  };

__device__ __forceinline__ v8f wmma16(v16h a, v16h b, v8f c) {
  return __builtin_amdgcn_wmma_f32_16x16x32_f16(false, a, false, b, (short)0, c,
                                                false, false);
}

__device__ __forceinline__ unsigned fkey(float v) {
  unsigned u = __float_as_uint(v);
  return (u & 0x80000000u) ? ~u : (u | 0x80000000u);
}
__device__ __forceinline__ float funkey(unsigned k) {
  return (k & 0x80000000u) ? __uint_as_float(k & 0x7fffffffu)
                           : __uint_as_float(~k);
}

// ---------------- small utility kernels ----------------
__global__ void zero_f32_kernel(float* p, int n) {
  int i = blockIdx.x * blockDim.x + threadIdx.x;
  if (i < n) p[i] = 0.0f;
}
__global__ void zero_u32_kernel(unsigned* p, int n) {
  int i = blockIdx.x * blockDim.x + threadIdx.x;
  if (i < n) p[i] = 0u;
}
__global__ void copy_points_kernel(const float* __restrict__ src, float* __restrict__ canvas) {
  int i = blockIdx.x * blockDim.x + threadIdx.x;
  if (i < BATCH * N0 * 3) {
    int b = i / (N0 * 3);
    int r = i % (N0 * 3);
    canvas[(size_t)b * CN * 3 + r] = src[i];
  }
}
__global__ void unkey_kernel(const unsigned* __restrict__ k, float* __restrict__ f, int n) {
  int i = blockIdx.x * blockDim.x + threadIdx.x;
  if (i < n) f[i] = funkey(k[i]);
}

// Pack W[K][N] (row-major f32) into fragment-ordered f16:
// out[((kt*NT+nt)*32 + lane)*16 + idx] = W[kt*32+frag_kk][nt*16+(lane&15)]
__global__ void pack_weights_kernel(const float* __restrict__ W, int K, int N, int NT,
                                    _Float16* __restrict__ out, int total) {
  int id = blockIdx.x * blockDim.x + threadIdx.x;
  if (id >= total) return;
  int idx  = id & 15;
  int lane = (id >> 4) & 31;
  int tile = id >> 9;
  int kt = tile / NT, nt = tile % NT;
  int k = kt * 32 + frag_kk(lane, idx);
  int n = nt * 16 + (lane & 15);
  float v = (k < K && n < N) ? W[(size_t)k * N + n] : 0.0f;
  out[id] = (_Float16)v;
}

// ---------------- encoder: fused 3->64->128->256 MLP + max-pool ----------
__global__ void encoder_kernel(const float* __restrict__ canvas,
                               const _Float16* __restrict__ pw1,
                               const _Float16* __restrict__ pw2,
                               const _Float16* __restrict__ pw3,
                               const float* __restrict__ b1,
                               const float* __restrict__ b2,
                               const float* __restrict__ b3,
                               unsigned* __restrict__ gkeys) {
  const int b    = blockIdx.y;
  const int wave = threadIdx.x >> 5;
  const int lane = threadIdx.x & 31;
  const int m0   = (blockIdx.x * 4 + wave) * 16;

  __shared__ _Float16 in0[4][16][32];
  __shared__ _Float16 act1[4][16][64];
  __shared__ _Float16 act2[4][16][128];
  __shared__ unsigned bmax[GD];

  for (int i = threadIdx.x; i < GD; i += blockDim.x) bmax[i] = 0u;
  for (int i = lane; i < 16 * 32; i += 32) {
    int m = i >> 5, k = i & 31;
    float v = (k < 3) ? canvas[((size_t)b * CN + (m0 + m)) * 3 + k] : 0.0f;
    in0[wave][m][k] = (_Float16)v;
  }
  __syncthreads();

  const int mrow  = lane & 15;
  const int mbase = (lane >> 4) * 8;

  // layer 1: K=32(pad from 3), N=64 (4 n-tiles)
  AFrag a0;
#pragma unroll
  for (int idx = 0; idx < 16; ++idx) a0.e[idx] = in0[wave][mrow][frag_kk(lane, idx)];
#pragma unroll
  for (int nt = 0; nt < 4; ++nt) {
    CFrag c; c.v = v8f{};
    v16h bf = *(const v16h*)(pw1 + (((size_t)nt) * 32 + lane) * 16);
    c.v = wmma16(a0.v, bf, c.v);
    int n = nt * 16 + mrow;
    float bias = b1[n];
#pragma unroll
    for (int j = 0; j < 8; ++j) {
      float v = c.e[j] + bias; v = v > 0.f ? v : 0.f;
      act1[wave][mbase + j][n] = (_Float16)v;
    }
  }
  __syncthreads();

  // layer 2: K=64 (2 k-tiles), N=128 (8 n-tiles)
  for (int nt = 0; nt < 8; ++nt) {
    CFrag c; c.v = v8f{};
#pragma unroll
    for (int kt = 0; kt < 2; ++kt) {
      AFrag a;
#pragma unroll
      for (int idx = 0; idx < 16; ++idx)
        a.e[idx] = act1[wave][mrow][kt * 32 + frag_kk(lane, idx)];
      v16h bf = *(const v16h*)(pw2 + (((size_t)kt * 8 + nt) * 32 + lane) * 16);
      c.v = wmma16(a.v, bf, c.v);
    }
    int n = nt * 16 + mrow;
    float bias = b2[n];
#pragma unroll
    for (int j = 0; j < 8; ++j) {
      float v = c.e[j] + bias; v = v > 0.f ? v : 0.f;
      act2[wave][mbase + j][n] = (_Float16)v;
    }
  }
  __syncthreads();

  // layer 3: K=128 (4 k-tiles), N=256 (16 n-tiles), no relu; block max-pool
  for (int nt = 0; nt < 16; ++nt) {
    CFrag c; c.v = v8f{};
#pragma unroll
    for (int kt = 0; kt < 4; ++kt) {
      AFrag a;
#pragma unroll
      for (int idx = 0; idx < 16; ++idx)
        a.e[idx] = act2[wave][mrow][kt * 32 + frag_kk(lane, idx)];
      v16h bf = *(const v16h*)(pw3 + (((size_t)kt * 16 + nt) * 32 + lane) * 16);
      c.v = wmma16(a.v, bf, c.v);
    }
    int n = nt * 16 + mrow;
    float bias = b3[n];
    float mx = -3.0e38f;
#pragma unroll
    for (int j = 0; j < 8; ++j) { float v = c.e[j] + bias; mx = v > mx ? v : mx; }
    atomicMax(&bmax[n], fkey(mx));
  }
  __syncthreads();
  for (int i = threadIdx.x; i < GD; i += blockDim.x)
    atomicMax(&gkeys[b * GD + i], bmax[i]);
}

// ---------------- attention: [xyz|gfeat|h](387)->128 relu ->1 ------------
__global__ void attention_kernel(const float* __restrict__ canvas,
                                 const float* __restrict__ gfeat,
                                 const float* __restrict__ hbuf,
                                 const _Float16* __restrict__ pwa,
                                 const float* __restrict__ ab1,
                                 const float* __restrict__ aw2,
                                 const float* __restrict__ ab2,
                                 float* __restrict__ scores) {
  const int b    = blockIdx.y;
  const int wave = threadIdx.x >> 5;
  const int lane = threadIdx.x & 31;
  const int m0   = (blockIdx.x * 4 + wave) * 16;

  __shared__ _Float16 gh[GD + HD];           // gfeat then h (batch-constant)
  __shared__ _Float16 xyz[4][16][4];
  __shared__ float    sc[4][16];
  __shared__ float    w2s[HD];

  for (int i = threadIdx.x; i < GD; i += blockDim.x) gh[i] = (_Float16)gfeat[b * GD + i];
  for (int i = threadIdx.x; i < HD; i += blockDim.x) {
    gh[GD + i] = (_Float16)hbuf[b * HD + i];
    w2s[i] = aw2[i];
  }
  for (int i = lane; i < 16 * 4; i += 32) {
    int m = i >> 2, k = i & 3;
    xyz[wave][m][k] = (_Float16)((k < 3) ? canvas[((size_t)b * CN + m0 + m) * 3 + k] : 0.0f);
  }
  if (lane < 16) sc[wave][lane] = 0.0f;
  __syncthreads();

  const int mrow  = lane & 15;
  const int mbase = (lane >> 4) * 8;

  for (int nt = 0; nt < 8; ++nt) {
    CFrag c; c.v = v8f{};
    for (int kt = 0; kt < CTX_KT; ++kt) {
      AFrag a;
#pragma unroll
      for (int idx = 0; idx < 16; ++idx) {
        int k = kt * 32 + frag_kk(lane, idx);
        _Float16 v;
        if (k < 3)          v = xyz[wave][mrow][k];
        else if (k < CTXK)  v = gh[k - 3];
        else                v = (_Float16)0.0f;
        a.e[idx] = v;
      }
      v16h bf = *(const v16h*)(pwa + (((size_t)kt * 8 + nt) * 32 + lane) * 16);
      c.v = wmma16(a.v, bf, c.v);
    }
    int n = nt * 16 + mrow;
    float bias = ab1[n], w2 = w2s[n];
#pragma unroll
    for (int j = 0; j < 8; ++j) {
      float v = c.e[j] + bias; v = v > 0.f ? v : 0.f;
      atomicAdd(&sc[wave][mbase + j], v * w2);
    }
  }
  __syncthreads();
  if (lane < 16) scores[(size_t)b * CN + m0 + lane] = sc[wave][lane] + ab2[0];
}

// ---------------- block reductions ----------------
__device__ float blk_sum(float v, float* sh) {
  int tid = threadIdx.x;
  sh[tid] = v; __syncthreads();
  for (int o = blockDim.x >> 1; o > 0; o >>= 1) {
    if (tid < o) sh[tid] += sh[tid + o];
    __syncthreads();
  }
  float r = sh[0]; __syncthreads();
  return r;
}
__device__ float blk_max(float v, float* sh) {
  int tid = threadIdx.x;
  sh[tid] = v; __syncthreads();
  for (int o = blockDim.x >> 1; o > 0; o >>= 1) {
    if (tid < o) sh[tid] = fmaxf(sh[tid], sh[tid + o]);
    __syncthreads();
  }
  float r = sh[0]; __syncthreads();
  return r;
}

// ---------------- softmax over points + attention center -----------------
__global__ void softmax_center_kernel(const float* __restrict__ canvas,
                                      const float* __restrict__ scores, int N,
                                      float* __restrict__ center) {
  const int b = blockIdx.x, tid = threadIdx.x;
  __shared__ float sh[256];
  float mx = -3.0e38f;
  for (int i = tid; i < N; i += blockDim.x)
    mx = fmaxf(mx, scores[(size_t)b * CN + i]);
  mx = blk_max(mx, sh);

  float se = 0.f, sx = 0.f, sy = 0.f, sz = 0.f;
  for (int i = tid; i < N; i += blockDim.x) {
    float e = __expf(scores[(size_t)b * CN + i] - mx);
    const float* p = &canvas[((size_t)b * CN + i) * 3];
    se += e; sx += e * p[0]; sy += e * p[1]; sz += e * p[2];
  }
  se = blk_sum(se, sh);
  sx = blk_sum(sx, sh);
  sy = blk_sum(sy, sh);
  sz = blk_sum(sz, sh);
  if (tid == 0) {
    center[b * 3 + 0] = sx / se;
    center[b * 3 + 1] = sy / se;
    center[b * 3 + 2] = sz / se;
  }
}

// ---------------- top-64 nearest + patch feat + assemble lstm_in ---------
__global__ void topk_lstmin_kernel(const float* __restrict__ canvas, int N,
                                   const float* __restrict__ center,
                                   const float* __restrict__ gfeat,
                                   float* __restrict__ lstm_in) {
  const int b = blockIdx.x, tid = threadIdx.x;
  __shared__ float rd[256];
  __shared__ int   ri[256];
  __shared__ float sel[3];
  __shared__ float lastd_s;
  __shared__ int   lasti_s;
  const float cx = center[b * 3], cy = center[b * 3 + 1], cz = center[b * 3 + 2];
  if (tid == 0) { sel[0] = sel[1] = sel[2] = 0.f; lastd_s = -1.f; lasti_s = -1; }
  __syncthreads();

  for (int it = 0; it < KSEL; ++it) {
    float ld = lastd_s; int li = lasti_s;
    float bd = 3.0e38f; int bi = 0x7fffffff;
    for (int i = tid; i < N; i += blockDim.x) {
      const float* p = &canvas[((size_t)b * CN + i) * 3];
      float dx = p[0] - cx, dy = p[1] - cy, dz = p[2] - cz;
      float d = dx * dx + dy * dy + dz * dz;
      bool after = (d > ld) || (d == ld && i > li);
      if (after && (d < bd || (d == bd && i < bi))) { bd = d; bi = i; }
    }
    rd[tid] = bd; ri[tid] = bi; __syncthreads();
    for (int o = blockDim.x >> 1; o > 0; o >>= 1) {
      if (tid < o) {
        if (rd[tid + o] < rd[tid] || (rd[tid + o] == rd[tid] && ri[tid + o] < ri[tid])) {
          rd[tid] = rd[tid + o]; ri[tid] = ri[tid + o];
        }
      }
      __syncthreads();
    }
    if (tid == 0) {
      int s = ri[0];
      const float* p = &canvas[((size_t)b * CN + s) * 3];
      sel[0] += p[0]; sel[1] += p[1]; sel[2] += p[2];
      lastd_s = rd[0]; lasti_s = s;
    }
    __syncthreads();
  }
  for (int i = tid; i < GD + 6; i += blockDim.x) {
    float v;
    if (i < GD)           v = gfeat[b * GD + i];
    else if (i < GD + 3)  v = center[b * 3 + (i - GD)];
    else                  v = sel[i - GD - 3] / (float)KSEL - center[b * 3 + (i - GD - 3)];
    lstm_in[b * (GD + 6) + i] = v;
  }
}

// ---------------- LSTM cell (one block per batch, 512 threads) -----------
__global__ void lstm_kernel(const float* __restrict__ lstm_in,
                            const float* __restrict__ wih,
                            const float* __restrict__ whh,
                            const float* __restrict__ bih,
                            const float* __restrict__ bhh,
                            float* __restrict__ h, float* __restrict__ c) {
  const int b = blockIdx.x, t = threadIdx.x;   // t in [0,512)
  __shared__ float g[4 * HD];
  float acc = bih[t] + bhh[t];
  for (int k = 0; k < GD + 6; ++k) acc += lstm_in[b * (GD + 6) + k] * wih[(size_t)k * 4 * HD + t];
  for (int k = 0; k < HD; ++k)     acc += h[b * HD + k] * whh[(size_t)k * 4 * HD + t];
  g[t] = acc; __syncthreads();
  if (t < HD) {
    float gi = g[t], gf = g[HD + t], gg = g[2 * HD + t], go = g[3 * HD + t];
    float si = 1.f / (1.f + __expf(-gi));
    float sf = 1.f / (1.f + __expf(-gf));
    float so = 1.f / (1.f + __expf(-go));
    float cn = sf * c[b * HD + t] + si * tanhf(gg);
    float hn = so * tanhf(cn);
    c[b * HD + t] = cn;
    h[b * HD + t] = hn;
  }
}

// ---------------- refinement MLP ----------------
__global__ void refine_hidden_kernel(const float* __restrict__ h,
                                     const float* __restrict__ w1,
                                     const float* __restrict__ b1,
                                     float* __restrict__ hr) {
  const int b = blockIdx.x, t = threadIdx.x;   // 128 threads
  float acc = b1[t];
  for (int k = 0; k < HD; ++k) acc += h[b * HD + k] * w1[(size_t)k * HD + t];
  hr[b * HD + t] = acc > 0.f ? acc : 0.f;
}

__global__ void refine_out_kernel(const float* __restrict__ hr,
                                  const float* __restrict__ w2,
                                  const float* __restrict__ b2,
                                  const float* __restrict__ center,
                                  float* __restrict__ canvas, int Nbase) {
  const int b = blockIdx.y;
  const int o = blockIdx.x * blockDim.x + threadIdx.x;   // [0, 3456)
  __shared__ float hs[HD];
  if (threadIdx.x < HD) hs[threadIdx.x] = hr[b * HD + threadIdx.x];
  __syncthreads();
  if (o < NEWP * 3) {
    float acc = b2[o];
    for (int k = 0; k < HD; ++k) acc += hs[k] * w2[(size_t)k * (NEWP * 3) + o];
    int p = o / 3, d = o % 3;
    canvas[((size_t)b * CN + Nbase + p) * 3 + d] = acc * 0.02f + center[b * 3 + d];
  }
}

// ---------------- chamfer: one direction, mean of row-mins ---------------
__global__ void chamfer_kernel(const float* __restrict__ X, int xoff, int NX, int xstride,
                               const float* __restrict__ Y, int yoff, int NY, int ystride,
                               float w, float* __restrict__ out) {
  const int id = blockIdx.x * blockDim.x + threadIdx.x;
  if (id >= BATCH * NX) return;
  const int b = id / NX, n = id % NX;
  const float* x  = &X[((size_t)b * xstride + xoff + n) * 3];
  const float x0 = x[0], x1 = x[1], x2 = x[2];
  const float* yb = &Y[((size_t)b * ystride + yoff) * 3];
  float best = 3.0e38f;
  for (int m = 0; m < NY; ++m) {
    float d0 = x0 - yb[m * 3 + 0];
    float d1 = x1 - yb[m * 3 + 1];
    float d2 = x2 - yb[m * 3 + 2];
    float d = d0 * d0 + d1 * d1 + d2 * d2;
    best = d < best ? d : best;
  }
  atomicAdd(out, w * best / (float)(BATCH * NX));
}

// ---------------- host launch ----------------
extern "C" void kernel_launch(void* const* d_in, const int* in_sizes, int n_in,
                              void* d_out, int out_size, void* d_ws, size_t ws_size,
                              hipStream_t stream) {
  const float* points  = (const float*)d_in[0];
  const float* gt      = (const float*)d_in[1];
  const float* enc_w1  = (const float*)d_in[2];
  const float* enc_b1  = (const float*)d_in[3];
  const float* enc_w2  = (const float*)d_in[4];
  const float* enc_b2  = (const float*)d_in[5];
  const float* enc_w3  = (const float*)d_in[6];
  const float* enc_b3  = (const float*)d_in[7];
  const float* att_w1  = (const float*)d_in[8];
  const float* att_b1  = (const float*)d_in[9];
  const float* att_w2  = (const float*)d_in[10];
  const float* att_b2  = (const float*)d_in[11];
  const float* lwih    = (const float*)d_in[12];
  const float* lwhh    = (const float*)d_in[13];
  const float* lbih    = (const float*)d_in[14];
  const float* lbhh    = (const float*)d_in[15];
  const float* ref_w1  = (const float*)d_in[16];
  const float* ref_b1  = (const float*)d_in[17];
  const float* ref_w2  = (const float*)d_in[18];
  const float* ref_b2  = (const float*)d_in[19];
  float* out = (float*)d_out;

  char* ws = (char*)d_ws;
  size_t off = 0;
  auto alloc = [&](size_t bytes) { size_t o = off; off += (bytes + 255) & ~(size_t)255; return o; };

  float*    canvas  = (float*)(ws + alloc((size_t)BATCH * CN * 3 * sizeof(float)));
  _Float16* pw1     = (_Float16*)(ws + alloc(2048  * sizeof(_Float16)));   // 1x4 tiles
  _Float16* pw2     = (_Float16*)(ws + alloc(8192  * sizeof(_Float16)));   // 2x8 tiles
  _Float16* pw3     = (_Float16*)(ws + alloc(32768 * sizeof(_Float16)));   // 4x16 tiles
  _Float16* pwa     = (_Float16*)(ws + alloc(53248 * sizeof(_Float16)));   // 13x8 tiles
  unsigned* gkeys   = (unsigned*)(ws + alloc(BATCH * GD * sizeof(unsigned)));
  float*    gfeat   = (float*)(ws + alloc(BATCH * GD * sizeof(float)));
  float*    hbuf    = (float*)(ws + alloc(BATCH * HD * sizeof(float)));
  float*    cbuf    = (float*)(ws + alloc(BATCH * HD * sizeof(float)));
  float*    scores  = (float*)(ws + alloc((size_t)BATCH * CN * sizeof(float)));
  float*    center  = (float*)(ws + alloc(BATCH * 3 * sizeof(float)));
  float*    lstm_in = (float*)(ws + alloc(BATCH * (GD + 6) * sizeof(float)));
  float*    hr      = (float*)(ws + alloc(BATCH * HD * sizeof(float)));

  // one-time (per launch) weight packing into WMMA fragment order
  pack_weights_kernel<<<(2048 + 255) / 256, 256, 0, stream>>>(enc_w1, 3, 64, 4, pw1, 2048);
  pack_weights_kernel<<<(8192 + 255) / 256, 256, 0, stream>>>(enc_w2, 64, 128, 8, pw2, 8192);
  pack_weights_kernel<<<(32768 + 255) / 256, 256, 0, stream>>>(enc_w3, 128, 256, 16, pw3, 32768);
  pack_weights_kernel<<<(53248 + 255) / 256, 256, 0, stream>>>(att_w1, CTXK, 128, 8, pwa, 53248);

  copy_points_kernel<<<(BATCH * N0 * 3 + 255) / 256, 256, 0, stream>>>(points, canvas);
  zero_f32_kernel<<<(BATCH * HD + 255) / 256, 256, 0, stream>>>(hbuf, BATCH * HD);
  zero_f32_kernel<<<(BATCH * HD + 255) / 256, 256, 0, stream>>>(cbuf, BATCH * HD);

  for (int step = 0; step < STEPS; ++step) {
    const int N = N0 + NEWP * step;   // multiple of 64 for every step
    zero_u32_kernel<<<(BATCH * GD + 255) / 256, 256, 0, stream>>>(gkeys, BATCH * GD);
    encoder_kernel<<<dim3(N / 64, BATCH), 128, 0, stream>>>(
        canvas, pw1, pw2, pw3, enc_b1, enc_b2, enc_b3, gkeys);
    unkey_kernel<<<(BATCH * GD + 255) / 256, 256, 0, stream>>>(gkeys, gfeat, BATCH * GD);
    attention_kernel<<<dim3(N / 64, BATCH), 128, 0, stream>>>(
        canvas, gfeat, hbuf, pwa, att_b1, att_w2, att_b2, scores);
    softmax_center_kernel<<<BATCH, 256, 0, stream>>>(canvas, scores, N, center);
    topk_lstmin_kernel<<<BATCH, 256, 0, stream>>>(canvas, N, center, gfeat, lstm_in);
    lstm_kernel<<<BATCH, 4 * HD, 0, stream>>>(lstm_in, lwih, lwhh, lbih, lbhh, hbuf, cbuf);
    refine_hidden_kernel<<<BATCH, HD, 0, stream>>>(hbuf, ref_w1, ref_b1, hr);
    refine_out_kernel<<<dim3((NEWP * 3 + 255) / 256, BATCH), 256, 0, stream>>>(
        hr, ref_w2, ref_b2, center, canvas, N);
  }

  zero_f32_kernel<<<1, 1, 0, stream>>>(out, 1);
  // 0.1 * chamfer(p_input, gt) + 1.0 * chamfer(p_new, gt)
  chamfer_kernel<<<(BATCH * N0 + 255) / 256, 256, 0, stream>>>(
      canvas, 0, N0, CN, gt, 0, N0, N0, 0.1f, out);
  chamfer_kernel<<<(BATCH * N0 + 255) / 256, 256, 0, stream>>>(
      gt, 0, N0, N0, canvas, 0, N0, CN, 0.1f, out);
  chamfer_kernel<<<(BATCH * NEWP * STEPS + 255) / 256, 256, 0, stream>>>(
      canvas, N0, NEWP * STEPS, CN, gt, 0, N0, N0, 1.0f, out);
  chamfer_kernel<<<(BATCH * N0 + 255) / 256, 256, 0, stream>>>(
      gt, 0, N0, N0, canvas, N0, NEWP * STEPS, CN, 1.0f, out);
}
